// FixedSimpleGNN_59098749993608
// MI455X (gfx1250) — matrix operand order repaired and checked by reading, not compiled, not run on hardware
//
#include <hip/hip_runtime.h>
#include <hip/hip_bf16.h>

#define F_IN 5
#define H 64
#define TPW 4   // tiles per wave

typedef __attribute__((ext_vector_type(16))) _Float16 v16h;
typedef __attribute__((ext_vector_type(8)))  _Float16 v8h;
typedef __attribute__((ext_vector_type(8)))  float    v8f;

__device__ __forceinline__ v8f wmma16(v16h a, v16h b, v8f c) {
  // D = A(16x32 f16) * B(32x16 f16) + C(16x16 f32)
  return __builtin_amdgcn_wmma_f32_16x16x32_f16(false, a, false, b, (short)0, c, false, false);
}

// Load one pre-swizzled weight A-fragment: lane's 16 f16 stored contiguously (32B aligned).
__device__ __forceinline__ v16h ldfrag(const _Float16* s, int f, int lane) {
  return *(const v16h*)(s + ((f * 32 + lane) << 4));
}

// Transposed MLP2 core: out holds (Hout)^T columns.
//   D1(64x16) = W1^T(64x128) @ X^T(128x16);  D2(64x16) = W2^T(64x64) @ relu(D1+b1)
// A (weights) from swizzled LDS; B (activations) read contiguously from row-major tiles:
//   B-fragment layout = (column n = lane&15, K = kc*32 + (lane>>4)*16 + i) -> 32B run.
// D layout: element r of v8f = (out-feature nt*16 + (lane>>4)*8 + r, edge/node lane&15).
__device__ __forceinline__ void mlp2_tile_T(const _Float16* aS, _Float16* hS, int lane,
                                            const _Float16* w1s, const float* b1s,
                                            const _Float16* w2s, const float* b2s,
                                            v8f out[4]) {
  const int n15 = lane & 15, kh = lane >> 4;
  // ---- layer 1 ----
  v16h xb[4];
#pragma unroll
  for (int kc = 0; kc < 4; ++kc)
    xb[kc] = *(const v16h*)(aS + n15 * 128 + kc * 32 + kh * 16);
#pragma unroll
  for (int nt = 0; nt < 4; ++nt) {
    v8f acc = {};
#pragma unroll
    for (int kc = 0; kc < 4; ++kc)
      acc = wmma16(ldfrag(w1s, kc * 4 + nt, lane), xb[kc], acc);
    const float4* bp = (const float4*)(b1s + nt * 16 + kh * 8);
    const float4 bA = bp[0], bB = bp[1];
    const float bv[8] = {bA.x, bA.y, bA.z, bA.w, bB.x, bB.y, bB.z, bB.w};
    v8h hh;
#pragma unroll
    for (int r = 0; r < 8; ++r) {
      const float v = acc[r] + bv[r];
      hh[r] = (_Float16)(v > 0.f ? v : 0.f);
    }
    *(v8h*)(hS + n15 * 64 + nt * 16 + kh * 8) = hh;   // contiguous 16B store
  }
  asm volatile("s_wait_dscnt 0" ::: "memory");        // cross-lane LDS RAW fence
  // ---- layer 2 ----
  v16h hb[2];
#pragma unroll
  for (int kc = 0; kc < 2; ++kc)
    hb[kc] = *(const v16h*)(hS + n15 * 64 + kc * 32 + kh * 16);
#pragma unroll
  for (int nt = 0; nt < 4; ++nt) {
    v8f acc = {};
#pragma unroll
    for (int kc = 0; kc < 2; ++kc)
      acc = wmma16(ldfrag(w2s, kc * 4 + nt, lane), hb[kc], acc);
    const float4* bp = (const float4*)(b2s + nt * 16 + kh * 8);
    const float4 bA = bp[0], bB = bp[1];
    const float bv[8] = {bA.x, bA.y, bA.z, bA.w, bB.x, bB.y, bB.z, bB.w};
#pragma unroll
    for (int r = 0; r < 8; ++r) {
      const float v = acc[r] + bv[r];
      out[nt][r] = v > 0.f ? v : 0.f;
    }
  }
}

// Swizzle W^T(Kdim x 64) into A-fragment order during f32->f16 conversion.
// frag f = kc*4+nt; within fragment: lane, element i:
//   m = nt*16 + (lane&15); k = kc*32 + (lane>>4)*16 + i; value = W[k*64+m]
__device__ __forceinline__ void stage_weights(const float* __restrict__ wg, _Float16* ws,
                                              int n_frag, int tid) {
  for (int idx = tid; idx < n_frag * 512; idx += 256) {
    const int f = idx >> 9;
    const int lane = (idx >> 4) & 31;
    const int i = idx & 15;
    const int kc = f >> 2, nt = f & 3;
    const int m = nt * 16 + (lane & 15);
    const int k = kc * 32 + ((lane >> 4) << 4) + i;
    ws[idx] = (_Float16)wg[k * 64 + m];
  }
}

// Stage one 64-float row as f16 into a row-major LDS tile (8 x 16B vector stores).
__device__ __forceinline__ void stage_row_f16(const float* __restrict__ row, _Float16* dst) {
  const float4* hp = (const float4*)row;
#pragma unroll
  for (int i = 0; i < 8; ++i) {
    const float4 va = hp[2 * i], vb = hp[2 * i + 1];
    v8h t;
    t[0] = (_Float16)va.x; t[1] = (_Float16)va.y; t[2] = (_Float16)va.z; t[3] = (_Float16)va.w;
    t[4] = (_Float16)vb.x; t[5] = (_Float16)vb.y; t[6] = (_Float16)vb.z; t[7] = (_Float16)vb.w;
    *(v8h*)(dst + i * 8) = t;
  }
}

// ---------- encoder: h = relu(relu(x@W1+b1)@W2+b2); F_IN=5 too skinny for WMMA ----------
__global__ __launch_bounds__(64) void gnn_encode_kernel(
    const float* __restrict__ x, const float* __restrict__ w1, const float* __restrict__ b1,
    const float* __restrict__ w2, const float* __restrict__ b2, float* __restrict__ h) {
  __shared__ float xs[F_IN];
  __shared__ float ts[H];
  const int node = blockIdx.x;
  const int j = threadIdx.x;
  if (j < F_IN) xs[j] = x[node * F_IN + j];
  __syncthreads();
  float t = b1[j];
#pragma unroll
  for (int k = 0; k < F_IN; ++k) t += xs[k] * w1[k * H + j];
  ts[j] = t > 0.f ? t : 0.f;
  __syncthreads();
  float o = b2[j];
  for (int k = 0; k < H; ++k) o += ts[k] * w2[k * H + j];
  h[(long)node * H + j] = o > 0.f ? o : 0.f;
}

// ---------- in-degree counts -> 1/max(count,1) ----------
__global__ void gnn_count_kernel(const int* __restrict__ ei, float* __restrict__ cnt, int E) {
  const int e = blockIdx.x * blockDim.x + threadIdx.x;
  if (e < E) atomicAdd(&cnt[ei[E + e]], 1.0f);
}
__global__ void gnn_invert_kernel(float* __restrict__ cnt, int n) {
  const int i = blockIdx.x * blockDim.x + threadIdx.x;
  if (i < n) {
    const float c = cnt[i];
    cnt[i] = 1.0f / (c > 1.0f ? c : 1.0f);
  }
}

// ---------- message kernel: 16-edge tiles, MLP2(concat(h[src],h[dst])) -> scaled atomic scatter ----------
__global__ __launch_bounds__(256) void gnn_msg_kernel(
    const float* __restrict__ h, const int* __restrict__ ei, const float* __restrict__ invden,
    const float* __restrict__ w1g, const float* __restrict__ b1g,
    const float* __restrict__ w2g, const float* __restrict__ b2g,
    float* __restrict__ agg, int n_edges) {
  __shared__ __align__(32) _Float16 w1s[16 * 512];
  __shared__ __align__(32) _Float16 w2s[8 * 512];
  __shared__ __align__(16) float b1s[64], b2s[64];
  __shared__ __align__(32) _Float16 aS[8][16 * 128];
  __shared__ __align__(32) _Float16 hS[8][16 * 64];

  stage_weights(w1g, w1s, 16, threadIdx.x);
  stage_weights(w2g, w2s, 8, threadIdx.x);
  if (threadIdx.x < 64) { b1s[threadIdx.x] = b1g[threadIdx.x]; b2s[threadIdx.x] = b2g[threadIdx.x]; }
  __syncthreads();

  const int lane = threadIdx.x & 31;
  const int wave = threadIdx.x >> 5;
  const int n15 = lane & 15, kh = lane >> 4;
  const int n_tiles = (n_edges + 15) >> 4;
  const int tile0 = (blockIdx.x * 8 + wave) * TPW;

  for (int t = 0; t < TPW; ++t) {
    const int tile = tile0 + t;
    if (tile >= n_tiles) break;
    const int e0 = tile << 4;
    int e = e0 + n15; if (e >= n_edges) e = n_edges - 1;
    const int dstn = ei[n_edges + e];
    const int node = kh ? dstn : ei[e];        // lane half 0: src row, half 1: dst row
    const float sc = invden[dstn];

    stage_row_f16(h + (long)node * H, &aS[wave][n15 * 128 + kh * 64]);
    asm volatile("s_wait_dscnt 0" ::: "memory");

    v8f out[4];
    mlp2_tile_T(aS[wave], hS[wave], lane, w1s, b1s, w2s, b2s, out);

    // lane owns edge n15's features [kh*8 + r] of each 16-chunk; fold 1/deg into scatter
    if (e0 + n15 < n_edges) {
      float* p = agg + (long)dstn * H + kh * 8;
#pragma unroll
      for (int nt = 0; nt < 4; ++nt)
#pragma unroll
        for (int r = 0; r < 8; ++r)
          atomicAdd(p + nt * 16 + r, out[nt][r] * sc);
    }
  }
}

// ---------- update kernel: 16-node tiles, h' = MLP2(concat(h, agg)) ----------
__global__ __launch_bounds__(256) void gnn_upd_kernel(
    const float* __restrict__ h, const float* __restrict__ agg,
    const float* __restrict__ w1g, const float* __restrict__ b1g,
    const float* __restrict__ w2g, const float* __restrict__ b2g,
    float* __restrict__ hn, int n_nodes) {
  __shared__ __align__(32) _Float16 w1s[16 * 512];
  __shared__ __align__(32) _Float16 w2s[8 * 512];
  __shared__ __align__(16) float b1s[64], b2s[64];
  __shared__ __align__(32) _Float16 aS[8][16 * 128];
  __shared__ __align__(32) _Float16 hS[8][16 * 64];

  stage_weights(w1g, w1s, 16, threadIdx.x);
  stage_weights(w2g, w2s, 8, threadIdx.x);
  if (threadIdx.x < 64) { b1s[threadIdx.x] = b1g[threadIdx.x]; b2s[threadIdx.x] = b2g[threadIdx.x]; }
  __syncthreads();

  const int lane = threadIdx.x & 31;
  const int wave = threadIdx.x >> 5;
  const int n15 = lane & 15, kh = lane >> 4;
  const int n_tiles = (n_nodes + 15) >> 4;
  const int tile0 = (blockIdx.x * 8 + wave) * TPW;

  for (int t = 0; t < TPW; ++t) {
    const int tile = tile0 + t;
    if (tile >= n_tiles) break;
    int node = (tile << 4) + n15; if (node >= n_nodes) node = n_nodes - 1;

    stage_row_f16((kh ? agg : h) + (long)node * H, &aS[wave][n15 * 128 + kh * 64]);
    asm volatile("s_wait_dscnt 0" ::: "memory");

    v8f out[4];
    mlp2_tile_T(aS[wave], hS[wave], lane, w1s, b1s, w2s, b2s, out);

    const int nd = (tile << 4) + n15;
    if (nd < n_nodes) {
      float* op = hn + (long)nd * H + kh * 8;
#pragma unroll
      for (int nt = 0; nt < 4; ++nt) {
        const float4 u = {out[nt][0], out[nt][1], out[nt][2], out[nt][3]};
        const float4 w = {out[nt][4], out[nt][5], out[nt][6], out[nt][7]};
        ((float4*)(op + nt * 16))[0] = u;
        ((float4*)(op + nt * 16))[1] = w;
      }
    }
  }
}

// ---------- readout: 2*pi*sigmoid(relu(h@W1+b1)@W2+b2), W2 is 64x3 ----------
__global__ __launch_bounds__(64) void gnn_out_kernel(
    const float* __restrict__ h, const float* __restrict__ w1, const float* __restrict__ b1,
    const float* __restrict__ w2, const float* __restrict__ b2, float* __restrict__ out) {
  __shared__ float hs[H];
  __shared__ float ts[H];
  const int node = blockIdx.x;
  const int j = threadIdx.x;
  hs[j] = h[(long)node * H + j];
  __syncthreads();
  float t = b1[j];
  for (int k = 0; k < H; ++k) t += hs[k] * w1[k * H + j];
  ts[j] = t > 0.f ? t : 0.f;
  __syncthreads();
  if (j < 3) {
    float p = b2[j];
    for (int k = 0; k < H; ++k) p += ts[k] * w2[k * 3 + j];
    out[(long)node * 3 + j] = 6.283185307179586f / (1.0f + __expf(-p));
  }
}

extern "C" void kernel_launch(void* const* d_in, const int* in_sizes, int n_in,
                              void* d_out, int out_size, void* d_ws, size_t ws_size,
                              hipStream_t stream) {
  const float* node_features = (const float*)d_in[0];
  const int*   edge_index    = (const int*)d_in[1];
  const float* enc_w1 = (const float*)d_in[2];
  const float* enc_b1 = (const float*)d_in[3];
  const float* enc_w2 = (const float*)d_in[4];
  const float* enc_b2 = (const float*)d_in[5];
  const float* msg_w1 = (const float*)d_in[6];
  const float* msg_b1 = (const float*)d_in[7];
  const float* msg_w2 = (const float*)d_in[8];
  const float* msg_b2 = (const float*)d_in[9];
  const float* upd_w1 = (const float*)d_in[10];
  const float* upd_b1 = (const float*)d_in[11];
  const float* upd_w2 = (const float*)d_in[12];
  const float* upd_b2 = (const float*)d_in[13];
  const float* out_w1 = (const float*)d_in[14];
  const float* out_b1 = (const float*)d_in[15];
  const float* out_w2 = (const float*)d_in[16];
  const float* out_b2 = (const float*)d_in[17];

  const int N = in_sizes[0] / F_IN;   // 100000
  const int E = in_sizes[1] / 2;      // 1000000

  // workspace layout: h ping-pong + agg + inv_degree (~77 MB)
  float* h0  = (float*)d_ws;
  float* h1  = h0 + (size_t)N * H;
  float* agg = h1 + (size_t)N * H;
  float* inv = agg + (size_t)N * H;

  gnn_encode_kernel<<<N, 64, 0, stream>>>(node_features, enc_w1, enc_b1, enc_w2, enc_b2, h0);

  (void)hipMemsetAsync(inv, 0, (size_t)N * sizeof(float), stream);
  gnn_count_kernel<<<(E + 255) / 256, 256, 0, stream>>>(edge_index, inv, E);
  gnn_invert_kernel<<<(N + 255) / 256, 256, 0, stream>>>(inv, N);

  const float* cur = h0;
  float* nxt = h1;
  const int e_tiles = (E + 15) / 16;
  const int n_tiles = (N + 15) / 16;
  const int msg_blocks = (e_tiles + 8 * TPW - 1) / (8 * TPW);
  const int upd_blocks = (n_tiles + 8 * TPW - 1) / (8 * TPW);
  for (int l = 0; l < 3; ++l) {
    (void)hipMemsetAsync(agg, 0, (size_t)N * H * sizeof(float), stream);
    gnn_msg_kernel<<<msg_blocks, 256, 0, stream>>>(
        cur, edge_index, inv,
        msg_w1 + (size_t)l * 2 * H * H, msg_b1 + (size_t)l * H,
        msg_w2 + (size_t)l * H * H,     msg_b2 + (size_t)l * H,
        agg, E);
    gnn_upd_kernel<<<upd_blocks, 256, 0, stream>>>(
        cur, agg,
        upd_w1 + (size_t)l * 2 * H * H, upd_b1 + (size_t)l * H,
        upd_w2 + (size_t)l * H * H,     upd_b2 + (size_t)l * H,
        nxt, N);
    float* t = (float*)cur; cur = nxt; nxt = t;
  }

  gnn_out_kernel<<<N, 64, 0, stream>>>(cur, out_w1, out_b1, out_w2, out_b2, (float*)d_out);
}
